// HybridGCN_29283087024198
// MI455X (gfx1250) — matrix operand order repaired and checked by reading, not compiled, not run on hardware
//
#include <hip/hip_runtime.h>
#include <stdint.h>

typedef unsigned short u16;
typedef __attribute__((ext_vector_type(16))) __bf16 v16bf;
typedef __attribute__((ext_vector_type(8)))  float  v8f;

#define NNODE  6144
#define NFEAT  512
#define NHID   64
#define NCLASS 16
#define NHEADS 2
#define NSPLIT 4
#define LRELU_ALPHA 0.2f
#define NEGBIG (-1.0e30f)

__device__ __forceinline__ u16 f2bf(float f) {
  union { float f; unsigned u; } t; t.f = f;
  unsigned u = t.u;
  unsigned r = u + 0x7FFFu + ((u >> 16) & 1u);   // round-to-nearest-even
  return (u16)(r >> 16);
}
__device__ __forceinline__ float bf2f(u16 b) {
  union { float f; unsigned u; } t; t.u = ((unsigned)b) << 16; return t.f;
}

union FragA {
  v16bf v;
  uint4 q[2];
  u16   s[16];
};

// ---------------- fp32 -> bf16 conversion ----------------
__global__ void cvt_f32_bf16(const float* __restrict__ in, u16* __restrict__ out, long n) {
  long i = (long)blockIdx.x * blockDim.x + threadIdx.x;
  if (i < n) out[i] = f2bf(in[i]);
}

// ---------------- bf16 WMMA GEMM with optional split-K ----------------
// C[M x NT*16] = A[M x K] * B[K x NT*16].  Block = 256 threads = 8 waves,
// each wave owns 16 rows.  grid.y = #K-splits; split s computes K-range
// [s*K/ns, (s+1)*K/ns) and writes to Cf + s*partStride (partStride=0 if ns==1).
// A frags load straight from row-major global; B tile staged transposed in LDS
// (pitch 40 u16), double buffered, one (column x 8 rows) chunk per thread.
template<int NT>
__global__ __launch_bounds__(256) void gemm_bf16(
    const u16* __restrict__ A, int lda,
    const u16* __restrict__ B, int ldb,
    float* __restrict__ Cf, u16* __restrict__ Cb, int ldc, int colOff,
    int K, int partStride)
{
  constexpr int NC = NT * 16;
  __shared__ __align__(16) u16 ldsB[2][NC * 40];
  const int tid  = threadIdx.x;
  const int lane = tid & 31;
  const int wave = tid >> 5;
  const int rowBase = blockIdx.x * 128 + wave * 16;
  const int kb  = (lane < 16) ? 0 : 8;
  const u16* arow = A + (unsigned)(rowBase + (lane & 15)) * (unsigned)lda + kb;

  const int ns    = (int)gridDim.y;
  const int kspan = K / ns;
  const int kbeg  = (int)blockIdx.y * kspan;
  const int kend  = kbeg + kspan;

  const bool doStage = (tid < NC * 4);
  const int  sc  = tid % NC;            // column
  const int  srb = (tid / NC) * 8;      // row base within 32-row tile
  const u16* bcol = B + sc;
  const unsigned ldbu = (unsigned)ldb;

  auto stage = [&](int buf, int k0) {
    if (doStage) {
      const u16* bp = bcol + (unsigned)(k0 + srb) * ldbu;
      unsigned d0 = (unsigned)bp[0]        | ((unsigned)bp[ldbu]     << 16);
      unsigned d1 = (unsigned)bp[2 * ldbu] | ((unsigned)bp[3 * ldbu] << 16);
      unsigned d2 = (unsigned)bp[4 * ldbu] | ((unsigned)bp[5 * ldbu] << 16);
      unsigned d3 = (unsigned)bp[6 * ldbu] | ((unsigned)bp[7 * ldbu] << 16);
      uint4 v; v.x = d0; v.y = d1; v.z = d2; v.w = d3;
      *(uint4*)&ldsB[buf][sc * 40 + srb] = v;
    }
  };

  v8f acc[NT] = {};
  stage(0, kbeg);
  int buf = 0;

  for (int k0 = kbeg; k0 < kend; k0 += 32) {
    __syncthreads();
    if (k0 + 32 < kend) stage(buf ^ 1, k0 + 32);

    FragA a;
    const u16* ap = arow + k0;
    a.q[0] = *(const uint4*)(ap);
    a.q[1] = *(const uint4*)(ap + 16);
    __builtin_prefetch(ap + 32, 0, 1);   // global_prefetch_b8: next K chunk

    FragA b[NT];
#pragma unroll
    for (int n = 0; n < NT; n++) {
      const u16* bp = &ldsB[buf][(n * 16 + (lane & 15)) * 40 + ((lane < 16) ? 0 : 8)];
      b[n].q[0] = *(const uint4*)(bp);
      b[n].q[1] = *(const uint4*)(bp + 16);
    }
#pragma unroll
    for (int n = 0; n < NT; n++)
      acc[n] = __builtin_amdgcn_wmma_f32_16x16x32_bf16(
          false, a.v, false, b[n].v, (short)0, acc[n], false, false);
    buf ^= 1;
  }

  float* cf = Cf ? Cf + (size_t)blockIdx.y * (size_t)partStride : nullptr;
#pragma unroll
  for (int n = 0; n < NT; n++) {
    int col = colOff + n * 16 + (lane & 15);
#pragma unroll
    for (int i = 0; i < 8; i++) {
      int r = rowBase + i + ((lane < 16) ? 0 : 8);
      float v = acc[n][i];
      if (cf) cf[(unsigned)r * (unsigned)ldc + col] = v;
      if (Cb) Cb[(unsigned)r * (unsigned)ldc + col] = f2bf(v);
    }
  }
}

// ---------------- deterministic 4-way partial reduction ----------------
__global__ void reduce4(const float* __restrict__ p, long stride,
                        float* __restrict__ out, long n) {
  long i = (long)blockIdx.x * blockDim.x + threadIdx.x;
  if (i < n)
    out[i] = (p[i] + p[i + stride]) + (p[i + 2 * stride] + p[i + 3 * stride]);
}

// ---------------- fused dense-GAT attention (flash-style, split-j, bf16 WMMA) ----------------
// grid.y = NSPLIT column-range splits.  Each split emits unnormalized row partials:
// accP[split][N][FC], mP[split][N], sP[split][N] (merged by attn_combine).
template<int NT>
__global__ __launch_bounds__(256) void gat_attn(
    const u16* __restrict__ adjb,
    const float* __restrict__ srcv, const float* __restrict__ dstv,
    const u16* __restrict__ Wh, int ldw, int wOff,
    float* __restrict__ accP, float* __restrict__ mP, float* __restrict__ sP)
{
  constexpr int FC = NT * 16;
  __shared__ __align__(16) u16 ldsW[2][FC * 40];
  __shared__ float ldsDst[2][32];
  const int tid  = threadIdx.x;
  const int lane = tid & 31;
  const int wave = tid >> 5;
  const int rowBase = blockIdx.x * 128 + wave * 16;
  const int row = rowBase + (lane & 15);
  const int kb  = (lane < 16) ? 0 : 8;
  const float sv = srcv[row];
  const u16* arow = adjb + (unsigned)row * (unsigned)NNODE + kb;

  const int jbeg = (int)blockIdx.y * (NNODE / NSPLIT);
  const int jend = jbeg + (NNODE / NSPLIT);

  const bool doStage = (tid < FC * 4);
  const int  sc  = tid % FC;
  const int  srb = (tid / FC) * 8;
  const u16* wcol = Wh + wOff + sc;
  const unsigned ldwu = (unsigned)ldw;

  auto stage = [&](int buf, int j0) {
    if (doStage) {
      const u16* bp = wcol + (unsigned)(j0 + srb) * ldwu;
      unsigned d0 = (unsigned)bp[0]        | ((unsigned)bp[ldwu]     << 16);
      unsigned d1 = (unsigned)bp[2 * ldwu] | ((unsigned)bp[3 * ldwu] << 16);
      unsigned d2 = (unsigned)bp[4 * ldwu] | ((unsigned)bp[5 * ldwu] << 16);
      unsigned d3 = (unsigned)bp[6 * ldwu] | ((unsigned)bp[7 * ldwu] << 16);
      uint4 v; v.x = d0; v.y = d1; v.z = d2; v.w = d3;
      *(uint4*)&ldsW[buf][sc * 40 + srb] = v;
    }
    if (tid < 32) ldsDst[buf][tid] = dstv[j0 + tid];
  };

  float m = NEGBIG, ssum = 0.0f;
  v8f acc[NT] = {};
  stage(0, jbeg);
  int buf = 0;

  for (int j0 = jbeg; j0 < jend; j0 += 32) {
    __syncthreads();
    if (j0 + 32 < jend) stage(buf ^ 1, j0 + 32);

    FragA adj;
    const u16* ap = arow + j0;
    adj.q[0] = *(const uint4*)(ap);
    adj.q[1] = *(const uint4*)(ap + 16);
    __builtin_prefetch(ap + 32, 0, 1);

    float e16[16];
    float tm = NEGBIG;
#pragma unroll
    for (int t = 0; t < 8; t++) {
      float x0 = sv + ldsDst[buf][kb + t];
      float l0 = x0 > 0.f ? x0 : LRELU_ALPHA * x0;
      e16[t] = adj.s[t] ? l0 : NEGBIG;
      float x1 = sv + ldsDst[buf][kb + 16 + t];
      float l1 = x1 > 0.f ? x1 : LRELU_ALPHA * x1;
      e16[8 + t] = adj.s[8 + t] ? l1 : NEGBIG;
      tm = fmaxf(tm, fmaxf(e16[t], e16[8 + t]));
    }
    tm = fmaxf(tm, __shfl_xor(tm, 16));          // combine lane pair -> full row max
    float mnew  = fmaxf(m, tm);
    float scale = __expf(m - mnew);

    FragA p;
    float ls = 0.f;
#pragma unroll
    for (int t = 0; t < 16; t++) {
      float pv = adj.s[t] ? __expf(e16[t] - mnew) : 0.0f;
      ls += pv;
      p.s[t] = f2bf(pv);
    }
    ls += __shfl_xor(ls, 16);
    ssum = ssum * scale + ls;
    m = mnew;

    // rescale accumulator rows: C-frag VGPR i holds row (i + 8*(lane>=16))
#pragma unroll
    for (int i = 0; i < 8; i++) {
      float scb = __shfl(scale, i + ((lane < 16) ? 0 : 8));
#pragma unroll
      for (int n = 0; n < NT; n++) acc[n][i] *= scb;
    }

    // WMMA in pairs: two B frags in flight to stagger dscnt waits
#pragma unroll
    for (int n0 = 0; n0 < NT; n0 += 2) {
      FragA b0, b1;
      {
        const u16* bp = &ldsW[buf][(n0 * 16 + (lane & 15)) * 40 + ((lane < 16) ? 0 : 8)];
        b0.q[0] = *(const uint4*)(bp);
        b0.q[1] = *(const uint4*)(bp + 16);
      }
      if (n0 + 1 < NT) {
        const u16* bp = &ldsW[buf][((n0 + 1) * 16 + (lane & 15)) * 40 + ((lane < 16) ? 0 : 8)];
        b1.q[0] = *(const uint4*)(bp);
        b1.q[1] = *(const uint4*)(bp + 16);
      }
      acc[n0] = __builtin_amdgcn_wmma_f32_16x16x32_bf16(
          false, p.v, false, b0.v, (short)0, acc[n0], false, false);
      if (n0 + 1 < NT)
        acc[n0 + 1] = __builtin_amdgcn_wmma_f32_16x16x32_bf16(
            false, p.v, false, b1.v, (short)0, acc[n0 + 1], false, false);
    }
    buf ^= 1;
  }

  // write unnormalized partials for this split
  float* accOut = accP + (size_t)blockIdx.y * ((size_t)NNODE * FC);
#pragma unroll
  for (int n = 0; n < NT; n++) {
    int col = n * 16 + (lane & 15);
#pragma unroll
    for (int i = 0; i < 8; i++) {
      int r = rowBase + i + ((lane < 16) ? 0 : 8);
      accOut[(unsigned)r * FC + col] = acc[n][i];
    }
  }
  if (lane < 16) {
    int r = rowBase + lane;
    mP[blockIdx.y * NNODE + r] = m;
    sP[blockIdx.y * NNODE + r] = ssum;
  }
}

// ---------------- merge flash-attention splits + normalize + ELU ----------------
template<int NT>
__global__ void attn_combine(const float* __restrict__ accP,
                             const float* __restrict__ mP, const float* __restrict__ sP,
                             float* __restrict__ Of, u16* __restrict__ Ob,
                             int ldo, int oOff) {
  constexpr int FC = NT * 16;
  int r = blockIdx.x * blockDim.x + threadIdx.x;
  if (r >= NNODE) return;
  float M = NEGBIG;
#pragma unroll
  for (int k = 0; k < NSPLIT; k++) M = fmaxf(M, mP[k * NNODE + r]);
  float w[NSPLIT], S = 0.f;
#pragma unroll
  for (int k = 0; k < NSPLIT; k++) {
    w[k] = __expf(mP[k * NNODE + r] - M);
    S += w[k] * sP[k * NNODE + r];
  }
  float inv = 1.0f / S;
  for (int c = 0; c < FC; c++) {
    float v = 0.f;
#pragma unroll
    for (int k = 0; k < NSPLIT; k++)
      v += w[k] * accP[(size_t)k * NNODE * FC + (unsigned)r * FC + c];
    v *= inv;
    v = v > 0.f ? v : (__expf(v) - 1.0f);       // ELU
    if (Of) Of[(unsigned)r * (unsigned)ldo + oOff + c] = v;
    if (Ob) Ob[(unsigned)r * (unsigned)ldo + oOff + c] = f2bf(v);
  }
}

// ---------------- src/dst attention logits ----------------
__global__ void srcdst_kernel(const u16* __restrict__ Wh, int ldw, int wOff, int F,
                              const float* __restrict__ a,
                              float* __restrict__ so, float* __restrict__ dsto, int n) {
  int r = blockIdx.x * blockDim.x + threadIdx.x;
  if (r >= n) return;
  float s1 = 0.f, s2 = 0.f;
  for (int j = 0; j < F; j++) {
    float w = bf2f(Wh[(unsigned)r * (unsigned)ldw + wOff + j]);
    s1 += w * a[j];
    s2 += w * a[F + j];
  }
  so[r] = s1; dsto[r] = s2;
}

// ---------------- relu(x + bias) -> bf16 ----------------
__global__ void bias_relu_bf16(const float* __restrict__ in, const float* __restrict__ b,
                               u16* __restrict__ out, int cols, long n) {
  long i = (long)blockIdx.x * blockDim.x + threadIdx.x;
  if (i < n) {
    float v = in[i] + b[i % cols];
    out[i] = f2bf(v > 0.f ? v : 0.f);
  }
}

// ---------------- concat + proj + log_softmax ----------------
__global__ void final_combine(const float* __restrict__ xgcn, const float* __restrict__ b2,
                              const float* __restrict__ xgat,
                              const float* __restrict__ pw, const float* __restrict__ pb,
                              float* __restrict__ out, int n) {
  int r = blockIdx.x * blockDim.x + threadIdx.x;
  if (r >= n) return;
  float c[32];
#pragma unroll
  for (int k = 0; k < 16; k++) {
    c[k]      = xgcn[(unsigned)r * 16 + k] + b2[k];
    c[16 + k] = xgat[(unsigned)r * 16 + k];
  }
  float lg[16], mx = -1e30f;
#pragma unroll
  for (int o = 0; o < 16; o++) {
    float a2 = pb[o];
#pragma unroll
    for (int k = 0; k < 32; k++) a2 += c[k] * pw[k * 16 + o];
    lg[o] = a2; mx = fmaxf(mx, a2);
  }
  float sum = 0.f;
#pragma unroll
  for (int o = 0; o < 16; o++) sum += __expf(lg[o] - mx);
  float l = __logf(sum);
#pragma unroll
  for (int o = 0; o < 16; o++) out[(unsigned)r * 16 + o] = lg[o] - mx - l;
}

extern "C" void kernel_launch(void* const* d_in, const int* in_sizes, int n_in,
                              void* d_out, int out_size, void* d_ws, size_t ws_size,
                              hipStream_t stream) {
  const float* x    = (const float*)d_in[0];
  const float* adj  = (const float*)d_in[1];
  const float* g1w  = (const float*)d_in[2];
  const float* g1b  = (const float*)d_in[3];
  const float* g2w  = (const float*)d_in[4];
  const float* g2b  = (const float*)d_in[5];
  const float* gatw = (const float*)d_in[6];
  const float* gata = (const float*)d_in[7];
  const float* gow  = (const float*)d_in[8];
  const float* goa  = (const float*)d_in[9];
  const float* pw   = (const float*)d_in[10];
  const float* pb   = (const float*)d_in[11];
  float* out = (float*)d_out;
  (void)in_sizes; (void)n_in; (void)out_size; (void)ws_size;

  const long N = NNODE;
  char* base = (char*)d_ws;
  size_t off = 0;
  auto alloc = [&](size_t bytes) -> void* {
    off = (off + 255) & ~(size_t)255;
    void* p = base + off;
    off += bytes;
    return p;
  };

  u16*   adjb  = (u16*)alloc((size_t)N * N * 2);
  u16*   xb    = (u16*)alloc((size_t)N * NFEAT * 2);
  u16*   w1b   = (u16*)alloc((size_t)NFEAT * NHID * 2);
  u16*   w2b   = (u16*)alloc((size_t)NHID * NCLASS * 2);
  u16*   gwb   = (u16*)alloc((size_t)NHEADS * NFEAT * NHID * 2);
  u16*   gob   = (u16*)alloc((size_t)NHEADS * NHID * NCLASS * 2);
  u16*   xw1b  = (u16*)alloc((size_t)N * NHID * 2);
  float* h1f   = (float*)alloc((size_t)N * NHID * 4);
  u16*   hb    = (u16*)alloc((size_t)N * NHID * 2);
  u16*   h2b   = (u16*)alloc((size_t)N * NCLASS * 2);
  float* xgcn  = (float*)alloc((size_t)N * NCLASS * 4);
  u16*   whb   = (u16*)alloc((size_t)N * NHEADS * NHID * 2);
  float* s0 = (float*)alloc(N * 4); float* d0 = (float*)alloc(N * 4);
  float* s1 = (float*)alloc(N * 4); float* d1 = (float*)alloc(N * 4);
  float* s2 = (float*)alloc(N * 4); float* d2 = (float*)alloc(N * 4);
  u16*   xgatb = (u16*)alloc((size_t)N * NHEADS * NHID * 2);
  u16*   whob  = (u16*)alloc((size_t)N * NCLASS * 2);
  float* xgatf = (float*)alloc((size_t)N * NCLASS * 4);
  // split partials
  float* partC = (float*)alloc((size_t)NSPLIT * N * NHID * 4);   // gemm split-K partials (>= N*16 case too)
  float* partA = (float*)alloc((size_t)NSPLIT * N * NHID * 4);   // attention acc partials
  float* partM = (float*)alloc((size_t)NSPLIT * N * 4);
  float* partS = (float*)alloc((size_t)NSPLIT * N * 4);

  dim3 blk(256);
  auto cvt = [&](const float* src, u16* dst, long n) {
    unsigned g = (unsigned)((n + 255) / 256);
    cvt_f32_bf16<<<dim3(g), blk, 0, stream>>>(src, dst, n);
  };
  cvt(adj,  adjb, N * N);
  cvt(x,    xb,   N * NFEAT);
  cvt(g1w,  w1b,  (long)NFEAT * NHID);
  cvt(g2w,  w2b,  (long)NHID * NCLASS);
  cvt(gatw, gwb,  (long)NHEADS * NFEAT * NHID);
  cvt(gow,  gob,  (long)NHEADS * NHID * NCLASS);

  dim3 gM1(NNODE / 128, 1);        // single-K-split gemms
  dim3 gMS(NNODE / 128, NSPLIT);   // split-K / split-j heavy passes
  dim3 gRow((unsigned)((N + 255) / 256));

  // ---- GCN branch ----
  gemm_bf16<4><<<gM1, blk, 0, stream>>>(xb, NFEAT, w1b, NHID, nullptr, xw1b, NHID, 0, NFEAT, 0);
  // H1 = adj @ XW1 (split-K partials -> reduce)
  gemm_bf16<4><<<gMS, blk, 0, stream>>>(adjb, NNODE, xw1b, NHID, partC, nullptr, NHID, 0, NNODE, (int)(N * NHID));
  reduce4<<<dim3((unsigned)((N * NHID + 255) / 256)), blk, 0, stream>>>(partC, N * NHID, h1f, N * NHID);
  bias_relu_bf16<<<dim3((unsigned)((N * NHID + 255) / 256)), blk, 0, stream>>>(h1f, g1b, hb, NHID, N * NHID);
  gemm_bf16<1><<<gM1, blk, 0, stream>>>(hb, NHID, w2b, NCLASS, nullptr, h2b, NCLASS, 0, NHID, 0);
  // xgcn = adj @ H2 (split-K partials -> reduce)
  gemm_bf16<1><<<gMS, blk, 0, stream>>>(adjb, NNODE, h2b, NCLASS, partC, nullptr, NCLASS, 0, NNODE, (int)(N * NCLASS));
  reduce4<<<dim3((unsigned)((N * NCLASS + 255) / 256)), blk, 0, stream>>>(partC, N * NCLASS, xgcn, N * NCLASS);

  // ---- GAT branch: Wh per head (concatenated into [N, 128]) ----
  gemm_bf16<4><<<gM1, blk, 0, stream>>>(xb, NFEAT, gwb,                      NHID, nullptr, whb, NHEADS * NHID, 0,    NFEAT, 0);
  gemm_bf16<4><<<gM1, blk, 0, stream>>>(xb, NFEAT, gwb + (long)NFEAT * NHID, NHID, nullptr, whb, NHEADS * NHID, NHID, NFEAT, 0);
  srcdst_kernel<<<gRow, blk, 0, stream>>>(whb, NHEADS * NHID, 0,    NHID, gata,            s0, d0, (int)N);
  srcdst_kernel<<<gRow, blk, 0, stream>>>(whb, NHEADS * NHID, NHID, NHID, gata + 2 * NHID, s1, d1, (int)N);

  gat_attn<4><<<gMS, blk, 0, stream>>>(adjb, s0, d0, whb, NHEADS * NHID, 0,    partA, partM, partS);
  attn_combine<4><<<gRow, blk, 0, stream>>>(partA, partM, partS, nullptr, xgatb, NHEADS * NHID, 0);
  gat_attn<4><<<gMS, blk, 0, stream>>>(adjb, s1, d1, whb, NHEADS * NHID, NHID, partA, partM, partS);
  attn_combine<4><<<gRow, blk, 0, stream>>>(partA, partM, partS, nullptr, xgatb, NHEADS * NHID, NHID);

  // ---- GAT output layer ----
  gemm_bf16<1><<<gM1, blk, 0, stream>>>(xgatb, NHEADS * NHID, gob, NCLASS, nullptr, whob, NCLASS, 0, NHEADS * NHID, 0);
  srcdst_kernel<<<gRow, blk, 0, stream>>>(whob, NCLASS, 0, NCLASS, goa, s2, d2, (int)N);
  gat_attn<1><<<gMS, blk, 0, stream>>>(adjb, s2, d2, whob, NCLASS, 0, partA, partM, partS);
  attn_combine<1><<<gRow, blk, 0, stream>>>(partA, partM, partS, xgatf, nullptr, NCLASS, 0);

  // ---- combine + projection + log_softmax ----
  final_combine<<<gRow, blk, 0, stream>>>(xgcn, g2b, xgatf, pw, pb, out, (int)N);
}
